// RNN_47794396070453
// MI455X (gfx1250) — matrix operand order repaired and checked by reading, not compiled, not run on hardware
//
#include <hip/hip_runtime.h>
#include <math.h>

typedef __attribute__((ext_vector_type(16))) _Float16 v16h;
typedef __attribute__((ext_vector_type(8)))  float    v8f;
typedef __attribute__((ext_vector_type(4)))  float    f32x4;
typedef __attribute__((ext_vector_type(4)))  unsigned int u32x4;

#define B_ 64
#define S_ 512
#define I_ 128
#define H_ 1024
#define O_ 64
#define R_ 4
#define ALPHA 0.1f
#define BT 16   // batch rows per workgroup

// Workgroup barrier that fences ONLY LDS (dscnt), not global stores.
#define LDS_BARRIER() \
  __asm__ volatile("s_wait_dscnt 0\n\ts_barrier_signal -1\n\ts_barrier_wait -1" ::: "memory")

// ---------------------------------------------------------------------------
// Kernel 1: hoist the step-invariant low-rank input factor
//   u[b,s,r] = sum_i x[b,s,i] * in_recv[i,r]        ([B,S,4], 512 KB in d_ws)
// ---------------------------------------------------------------------------
__global__ void rnn_input_lowrank(const float* __restrict__ x,
                                  const float* __restrict__ in_recv,
                                  float* __restrict__ u) {
  int t = blockIdx.x * blockDim.x + threadIdx.x;   // flat (b,s)
  if (t >= B_ * S_) return;
  const float* xp = x + (size_t)t * I_;
  float a0 = 0.f, a1 = 0.f, a2 = 0.f, a3 = 0.f;
  for (int i = 0; i < I_; ++i) {
    float xv = xp[i];
    const float* rr = in_recv + i * R_;
    a0 += xv * rr[0]; a1 += xv * rr[1]; a2 += xv * rr[2]; a3 += xv * rr[3];
  }
  float* up = u + (size_t)t * R_;
  up[0] = a0; up[1] = a1; up[2] = a2; up[3] = a3;
}

// ---------------------------------------------------------------------------
// Kernel 2: sequential scan. One workgroup per 16 batch rows (4 WGs).
// 1024 threads = 32 waves. h-state in registers (16 f32 per thread).
// Readout weights pre-converted to f16, resident in LDS in exact WMMA
// B-fragment order. Wave w owns ntile = w&3, kgroup = w>>2.
// ---------------------------------------------------------------------------
__global__ void __launch_bounds__(1024, 1)
rnn_scan(const float* __restrict__ u,        // [B,S,R]
         const float* __restrict__ iproj_g,  // [H,R]  input_projective
         const float* __restrict__ rproj_g,  // [H,R]  reccurent_projective
         const float* __restrict__ rrecv_g,  // [H,R]  reccurent_receptive
         const float* __restrict__ w_g,      // [O,H]  readout_w
         const float* __restrict__ bias_g,   // [O]
         float* __restrict__ out) {          // [B,S,O]
  __shared__ __align__(16) _Float16 h16[BT * H_];        // 32 KB f16 mirror of h (WMMA A)
  __shared__ __align__(16) float rrecv[H_ * R_];         // 16 KB
  __shared__ __align__(16) float rproj[H_ * R_];         // 16 KB
  __shared__ __align__(16) float iproj[H_ * R_];         // 16 KB
  __shared__ __align__(16) u32x4 w16[32][4][2][32];      // 128 KB f16 W in B-frag order
  __shared__ __align__(16) float partial[32][256];       // 32 KB per-wave C tiles
  __shared__ __align__(16) float p_part[32][4];          // per-wave rank-4 partials
  __shared__ float bias_s[O_];

  const int tid    = threadIdx.x;
  const int b0     = blockIdx.x * BT;
  const int wave   = tid >> 5;
  const int lane   = tid & 31;
  const int ntile  = wave & 3;    // which 16 output columns
  const int kgroup = wave >> 2;   // which 128-wide K slice

  // ---- preload projection tables + bias into LDS
  for (int idx = tid; idx < H_ * R_; idx += 1024) {
    rrecv[idx] = rrecv_g[idx];
    rproj[idx] = rproj_g[idx];
    iproj[idx] = iproj_g[idx];
  }
  if (tid < O_) bias_s[tid] = bias_g[tid];

  // ---- convert readout_w -> f16 B-fragments in LDS (once).
  {
    const int n     = ntile * 16 + (lane & 15);
    const int khalf = (lane >> 4) * 16;
    for (int c = 0; c < 4; ++c) {
      const int kk = kgroup * 128 + c * 32 + khalf;
      const float* wp = w_g + n * H_ + kk;
      union { v16h v; u32x4 q[2]; } bh;
#pragma unroll
      for (int j = 0; j < 16; ++j) bh.v[j] = (_Float16)wp[j];
      w16[wave][c][0][lane] = bh.q[0];
      w16[wave][c][1][lane] = bh.q[1];
    }
  }

  // ---- h state in registers: thread owns batch row hb, H indices hbase+64*j
  const int hb    = tid >> 6;   // 0..15  (waves 2*hb and 2*hb+1 share this row)
  const int hbase = tid & 63;
  float hreg[16];
#pragma unroll
  for (int j = 0; j < 16; ++j) hreg[j] = 0.f;

  // prefetch u for step 0 (subsequent steps prefetched one step ahead)
  const float* ubase = u + (size_t)(b0 + hb) * S_ * R_;
  f32x4 unext = *(const f32x4*)(ubase);

  __syncthreads();

  for (int s = 0; s < S_; ++s) {
    // ---- phase 1: per-wave partial of p[b][r] = sum_h h[b][h]*rrecv[h][r]
    {
      float a0 = 0.f, a1 = 0.f, a2 = 0.f, a3 = 0.f;
#pragma unroll
      for (int j = 0; j < 16; ++j) {
        const int h = hbase + 64 * j;
        const float hv = hreg[j];
        const f32x4 rr = *(const f32x4*)&rrecv[h * R_];
        a0 += hv * rr.x; a1 += hv * rr.y; a2 += hv * rr.z; a3 += hv * rr.w;
      }
#pragma unroll
      for (int off = 16; off > 0; off >>= 1) {
        a0 += __shfl_xor(a0, off, 32);
        a1 += __shfl_xor(a1, off, 32);
        a2 += __shfl_xor(a2, off, 32);
        a3 += __shfl_xor(a3, off, 32);
      }
      if (lane == 0) {
        f32x4 pv; pv.x = a0; pv.y = a1; pv.z = a2; pv.w = a3;
        *(f32x4*)&p_part[wave][0] = pv;
      }
    }
    LDS_BARRIER();

    // ---- phase 2: Euler update (pure f32) + f16 mirror for WMMA
    {
      const f32x4 pA = *(const f32x4*)&p_part[2 * hb][0];
      const f32x4 pB = *(const f32x4*)&p_part[2 * hb + 1][0];
      const f32x4 p  = pA + pB;
      const f32x4 uc = unext;
      if (s + 1 < S_)                       // prefetch next step's input drive;
        unext = *(const f32x4*)(ubase + (size_t)(s + 1) * R_);  // hidden by 3/3b/1
#pragma unroll
      for (int j = 0; j < 16; ++j) {
        const int h = hbase + 64 * j;
        const f32x4 rp = *(const f32x4*)&rproj[h * R_];
        const f32x4 ip = *(const f32x4*)&iproj[h * R_];
        float rin = p.x * rp.x + p.y * rp.y + p.z * rp.z + p.w * rp.w
                  + uc.x * ip.x + uc.y * ip.y + uc.z * ip.z + uc.w * ip.w;
#if __has_builtin(__builtin_amdgcn_tanhf)
        float th = __builtin_amdgcn_tanhf(rin);   // v_tanh_f32
#else
        float th = tanhf(rin);
#endif
        float hn = hreg[j] + ALPHA * (th - hreg[j]);
        hreg[j] = hn;
        h16[hb * H_ + h] = (_Float16)hn;
      }
    }
    LDS_BARRIER();

    // ---- phase 3: readout GEMM tile via WMMA, double-buffered operand loads
    // so ds_load_b128 latency overlaps the previous v_wmma.
    {
      const int m   = lane & 15;        // batch row (A) == column N (C)
      const int hi8 = (lane >> 4) * 8;
      union AB { v16h v; u32x4 q[2]; };
      AB aA, bA, aB, bB;

      // chunk 0 + chunk 1 loads in flight
      {
        const int k0 = kgroup * 128;
        aA.q[0] = *(const u32x4*)&h16[m * H_ + k0 + 2 * hi8];
        aA.q[1] = *(const u32x4*)&h16[m * H_ + k0 + 8 + 2 * hi8];
        bA.q[0] = w16[wave][0][0][lane];
        bA.q[1] = w16[wave][0][1][lane];
        aB.q[0] = *(const u32x4*)&h16[m * H_ + k0 + 32 + 2 * hi8];
        aB.q[1] = *(const u32x4*)&h16[m * H_ + k0 + 40 + 2 * hi8];
        bB.q[0] = w16[wave][1][0][lane];
        bB.q[1] = w16[wave][1][1][lane];
      }
      v8f acc = {};
      acc = __builtin_amdgcn_wmma_f32_16x16x32_f16(
          false, aA.v, false, bA.v, (short)0, acc, false, false);
      {
        const int k0 = kgroup * 128 + 64;
        aA.q[0] = *(const u32x4*)&h16[m * H_ + k0 + 2 * hi8];
        aA.q[1] = *(const u32x4*)&h16[m * H_ + k0 + 8 + 2 * hi8];
        bA.q[0] = w16[wave][2][0][lane];
        bA.q[1] = w16[wave][2][1][lane];
      }
      acc = __builtin_amdgcn_wmma_f32_16x16x32_f16(
          false, aB.v, false, bB.v, (short)0, acc, false, false);
      {
        const int k0 = kgroup * 128 + 96;
        aB.q[0] = *(const u32x4*)&h16[m * H_ + k0 + 2 * hi8];
        aB.q[1] = *(const u32x4*)&h16[m * H_ + k0 + 8 + 2 * hi8];
        bB.q[0] = w16[wave][3][0][lane];
        bB.q[1] = w16[wave][3][1][lane];
      }
      acc = __builtin_amdgcn_wmma_f32_16x16x32_f16(
          false, aA.v, false, bA.v, (short)0, acc, false, false);
      acc = __builtin_amdgcn_wmma_f32_16x16x32_f16(
          false, aB.v, false, bB.v, (short)0, acc, false, false);

      // C layout: VGPR v -> row M = v (+8 for lanes 16..31), col N = lane&15
#pragma unroll
      for (int v = 0; v < 8; ++v)
        partial[wave][(v + hi8) * 16 + m] = acc[v];
    }
    LDS_BARRIER();

    // ---- phase 3b: sum partials over the 8 K-groups (independent loads +
    // tree sum for ILP), add bias, store out. No trailing barrier needed.
    {
      const int nt  = tid >> 8;         // output n-tile
      const int m   = (tid >> 4) & 15;  // batch row
      const int n   = tid & 15;         // col within tile
      const int idx = m * 16 + n;
      const float q0 = partial[0 * 4 + nt][idx];
      const float q1 = partial[1 * 4 + nt][idx];
      const float q2 = partial[2 * 4 + nt][idx];
      const float q3 = partial[3 * 4 + nt][idx];
      const float q4 = partial[4 * 4 + nt][idx];
      const float q5 = partial[5 * 4 + nt][idx];
      const float q6 = partial[6 * 4 + nt][idx];
      const float q7 = partial[7 * 4 + nt][idx];
      const int o = nt * 16 + n;
      const float sum = ((q0 + q1) + (q2 + q3)) + ((q4 + q5) + (q6 + q7)) + bias_s[o];
      out[((size_t)(b0 + m) * S_ + s) * O_ + o] = sum;
    }
  }
}

// ---------------------------------------------------------------------------
extern "C" void kernel_launch(void* const* d_in, const int* in_sizes, int n_in,
                              void* d_out, int out_size, void* d_ws, size_t ws_size,
                              hipStream_t stream) {
  const float* x     = (const float*)d_in[0];  // [B,S,I]
  const float* iproj = (const float*)d_in[1];  // [H,R]
  const float* irecv = (const float*)d_in[2];  // [I,R]
  const float* rproj = (const float*)d_in[3];  // [H,R]
  const float* rrecv = (const float*)d_in[4];  // [H,R]
  const float* w     = (const float*)d_in[5];  // [O,H]
  const float* bias  = (const float*)d_in[6];  // [O]
  float* out = (float*)d_out;                  // [B,S,O]
  float* u   = (float*)d_ws;                   // [B,S,R] = 512 KB scratch

  rnn_input_lowrank<<<(B_ * S_ + 255) / 256, 256, 0, stream>>>(x, irecv, u);
  rnn_scan<<<B_ / BT, 1024, 0, stream>>>(u, iproj, rproj, rrecv, w, bias, out);
}
